// MoHAttention_19447611916393
// MI455X (gfx1250) — matrix operand order repaired and checked by reading, not compiled, not run on hardware
//
#include <hip/hip_runtime.h>

#define DIM      768
#define NH       12
#define HD       64
#define SEQ      2048
#define BATCH    4
#define NTOK     (BATCH * SEQ)   // 8192
#define E_ROUTED 10
#define LN_EPS   1e-5f

typedef __attribute__((ext_vector_type(16))) _Float16 v16h;
typedef __attribute__((ext_vector_type(8)))  _Float16 v8h;
typedef __attribute__((ext_vector_type(8)))  float    v8f;
typedef __attribute__((ext_vector_type(4)))  unsigned int u32x4;
typedef __attribute__((ext_vector_type(8)))  int      i32x8;
typedef __attribute__((ext_vector_type(4)))  int      i32x4;

// ---------------------------------------------------------------------------
// WMMA fragment helpers (wave32; contiguous k order, used alike for A and B)
// A 16x32 f16: lane m=L%16, half=L/16. a[i]=A[m][kb+16*half+i]
//              (one contiguous 32B chunk)
// B 32x16 f16 from k-major storage BT[n][k]: lane n=L%16, kh=L/16,
//              b[i]=BT[n][kb+16*kh+i]             (one contiguous 32B chunk)
// ---------------------------------------------------------------------------
__device__ __forceinline__ v16h frag_cat(v8h lo, v8h hi) {
  v16h a;
#pragma unroll
  for (int i = 0; i < 8; ++i) { a[i] = lo[i]; a[i + 8] = hi[i]; }
  return a;
}
__device__ __forceinline__ v16h load_a_frag(const _Float16* row, int kb, int half) {
  const _Float16* p = row + kb;
  return frag_cat(*(const v8h*)(p + 16 * half), *(const v8h*)(p + 16 * half + 8));
}
__device__ __forceinline__ v16h load_b_frag(const _Float16* row, int kb, int half) {
  const _Float16* p = row + kb + 16 * half;
  return frag_cat(*(const v8h*)(p), *(const v8h*)(p + 8));
}
__device__ __forceinline__ v8f wmma16(v16h a, v16h b, v8f c) {
  return __builtin_amdgcn_wmma_f32_16x16x32_f16(false, a, false, b,
                                                (short)0, c, false, false);
}
__device__ __forceinline__ float red_max16(float v) {
#pragma unroll
  for (int off = 1; off < 16; off <<= 1) v = fmaxf(v, __shfl_xor(v, off, 32));
  return v;
}
__device__ __forceinline__ float red_sum16(float v) {
#pragma unroll
  for (int off = 1; off < 16; off <<= 1) v += __shfl_xor(v, off, 32);
  return v;
}

// ---------------------------------------------------------------------------
// Tensor Data Mover: 2-D tile load (f16 elements) Global -> LDS.
// D# per CDNA5 ISA: group0 = {count/lds/global/type},
// group1 = {data_size, tensor dims, tile dims, dim0 stride}. 2-D => groups
// 2/3 zero. LDS byte offset = low 32 bits of the generic pointer (ISA §10.2).
// Tracked by TENSORcnt.
// ---------------------------------------------------------------------------
__device__ __forceinline__ unsigned lds_off(const void* p) {
  return (unsigned)(unsigned long long)p;
}

__device__ __forceinline__ void tdm_load_2d_f16(unsigned lds_byte_off,
                                                const void* gptr,
                                                unsigned tile_d0,   // contiguous elems
                                                unsigned tile_d1,   // rows
                                                unsigned row_stride_elems) {
  unsigned long long ga = (unsigned long long)gptr;
  u32x4 g0;
  g0[0] = 1u;                                               // count=1, user D#
  g0[1] = lds_byte_off;                                     // lds_addr
  g0[2] = (unsigned)(ga & 0xFFFFFFFFu);                     // global_addr[31:0]
  g0[3] = (unsigned)((ga >> 32) & 0x01FFFFFFu) | (2u << 30);// addr[56:32] | type=2
  i32x8 g1;
  g1[0] = (int)(1u << 16);                                  // data_size=1 (2 bytes)
  g1[1] = (int)((tile_d0 & 0xFFFFu) << 16);                 // tensor_dim0[15:0]
  g1[2] = (int)(((tile_d0 >> 16) & 0xFFFFu) |               // tensor_dim0[31:16]
                ((tile_d1 & 0xFFFFu) << 16));               // tensor_dim1[15:0]
  g1[3] = (int)(((tile_d1 >> 16) & 0xFFFFu) |               // tensor_dim1[31:16]
                ((tile_d0 & 0xFFFFu) << 16));               // tile_dim0
  g1[4] = (int)(tile_d1 & 0xFFFFu);                         // tile_dim1 (tile_dim2=0)
  g1[5] = (int)row_stride_elems;                            // tensor_dim0_stride[31:0]
  g1[6] = 0;
  g1[7] = 0;
  i32x4 g2 = {0, 0, 0, 0};
  i32x4 g3 = {0, 0, 0, 0};
#if __has_include(<hip/amd_detail/amd_gfx1250_TDM.h>)
  i32x8 g4 = {0, 0, 0, 0, 0, 0, 0, 0};
  __builtin_amdgcn_tensor_load_to_lds(g0, g1, g2, g3, g4, 0);
#else
  __builtin_amdgcn_tensor_load_to_lds(g0, g1, g2, g3, 0);
#endif
}

// ---------------------------------------------------------------------------
// Prep kernels
// ---------------------------------------------------------------------------
__global__ __launch_bounds__(256) void k_cvt_x(const float* __restrict__ x,
                                               _Float16* __restrict__ xh, int n) {
  int i = blockIdx.x * blockDim.x + threadIdx.x;
  if (i < n) xh[i] = (_Float16)x[i];
}

// w stored [in][out]; produce wt[out][in] in f16 (k-major for B fragments)
__global__ __launch_bounds__(256) void k_transpose_h(const float* __restrict__ w,
                                                     _Float16* __restrict__ wt,
                                                     int in_dim, int out_dim) {
  int i = blockIdx.x * blockDim.x + threadIdx.x;
  if (i >= in_dim * out_dim) return;
  int o = i / in_dim, k = i - o * in_dim;
  wt[(size_t)o * in_dim + k] = (_Float16)w[(size_t)k * out_dim + o];
}

// ---------------------------------------------------------------------------
// MoH gating: one wave per token -> masked_gates[t][12].
// ---------------------------------------------------------------------------
__global__ __launch_bounds__(256) void k_gating(const float* __restrict__ x,
                                                const float* __restrict__ wg,
                                                const float* __restrict__ wg0,
                                                const float* __restrict__ wg1,
                                                float* __restrict__ gout) {
  int wave = (int)((blockIdx.x * blockDim.x + threadIdx.x) >> 5);
  int lane = threadIdx.x & 31;
  const float* xr = x + (size_t)wave * DIM;
  float acc[14];
#pragma unroll
  for (int e = 0; e < 14; ++e) acc[e] = 0.f;
  for (int k = lane; k < DIM; k += 32) {
    float xv = xr[k];
#pragma unroll
    for (int e = 0; e < E_ROUTED; ++e) acc[e] += xv * wg[k * E_ROUTED + e];
    acc[10] += xv * wg0[k * 2 + 0];
    acc[11] += xv * wg0[k * 2 + 1];
    acc[12] += xv * wg1[k * 2 + 0];
    acc[13] += xv * wg1[k * 2 + 1];
  }
#pragma unroll
  for (int e = 0; e < 14; ++e) {
#pragma unroll
    for (int off = 16; off > 0; off >>= 1) acc[e] += __shfl_xor(acc[e], off, 32);
  }
  if (lane == 0) {
    float g[E_ROUTED];
    float mx = acc[0];
#pragma unroll
    for (int e = 1; e < E_ROUTED; ++e) mx = fmaxf(mx, acc[e]);
    float s = 0.f;
#pragma unroll
    for (int e = 0; e < E_ROUTED; ++e) { g[e] = __expf(acc[e] - mx); s += g[e]; }
    float inv = 1.f / s;
#pragma unroll
    for (int e = 0; e < E_ROUTED; ++e) g[e] *= inv;
    int i1 = 0;
#pragma unroll
    for (int e = 1; e < E_ROUTED; ++e) if (g[e] > g[i1]) i1 = e;
    int i2 = (i1 == 0) ? 1 : 0;
#pragma unroll
    for (int e = 0; e < E_ROUTED; ++e) if (e != i1 && g[e] > g[i2]) i2 = e;
    float denom = fmaxf(g[i1] + g[i2], 1.1920929e-7f);
    float m0 = fmaxf(acc[10], acc[11]);
    float e0 = __expf(acc[10] - m0), e1 = __expf(acc[11] - m0);
    float w00 = 2.f * e0 / (e0 + e1), w01 = 2.f * e1 / (e0 + e1);
    float m1 = fmaxf(acc[12], acc[13]);
    float f0 = __expf(acc[12] - m1), f1 = __expf(acc[13] - m1);
    float fs = 1.f / (f0 + f1);
    float* go = gout + (size_t)wave * NH;
    go[0] = w00 * 2.f * f0 * fs;
    go[1] = w00 * 2.f * f1 * fs;
#pragma unroll
    for (int e = 0; e < E_ROUTED; ++e)
      go[2 + e] = (e == i1 || e == i2) ? w01 * (g[e] / denom * 2.f) : 0.f;
  }
}

// ---------------------------------------------------------------------------
// QKV GEMM + fused q/k LayerNorm + V transpose.
// Block = 128 thr (4 waves) = 64 tokens x one 64-col head strip.
// Wave w: 16 tokens x 64 cols (A fragment reused across 4 WMMAs).
// Weight panel (64 cols x 32 k) TDM-staged into LDS, double buffered.
// ---------------------------------------------------------------------------
__global__ __launch_bounds__(128) void k_qkv(const _Float16* __restrict__ xh,
                                             const _Float16* __restrict__ wT,
                                             const float* __restrict__ bias,
                                             const float* __restrict__ qnw,
                                             const float* __restrict__ qnb,
                                             const float* __restrict__ knw,
                                             const float* __restrict__ knb,
                                             _Float16* __restrict__ qbuf,
                                             _Float16* __restrict__ kbuf,
                                             _Float16* __restrict__ vtbuf) {
  __shared__ __align__(16) _Float16 sB[2][64 * 32];  // [col][k] panel
  __shared__ float tile[64][65];
  __shared__ float s_mean[64], s_rstd[64];
  int tt = blockIdx.x;          // 64-token tile
  int strip = blockIdx.y;       // 0..35
  int w = threadIdx.x >> 5, lane = threadIdx.x & 31;
  int m = lane & 15, half = lane >> 4;
  int s = strip / NH, h = strip - s * NH;

  const _Float16* arow = xh + (size_t)(tt * 64 + w * 16 + m) * DIM;
  const _Float16* wbase = wT + (size_t)(strip * 64) * DIM;
  unsigned l0 = lds_off(&sB[0][0]), l1 = lds_off(&sB[1][0]);

  v8f c0 = {0.f,0.f,0.f,0.f,0.f,0.f,0.f,0.f};
  v8f c1 = c0, c2 = c0, c3 = c0;

  if (w == 0) tdm_load_2d_f16(l0, wbase, 32, 64, DIM);
  int cur = 0;
  for (int kb = 0; kb < DIM; kb += 32) {
    if (w == 0) {
      if (kb + 32 < DIM) {
        tdm_load_2d_f16(cur ? l0 : l1, wbase + kb + 32, 32, 64, DIM);
        __builtin_amdgcn_s_wait_tensorcnt(1);
      } else {
        __builtin_amdgcn_s_wait_tensorcnt(0);
      }
    }
    __syncthreads();
    const _Float16* bp = &sB[cur][0];
    v16h a = load_a_frag(arow, kb, half);
    c0 = wmma16(a, load_b_frag(bp + (size_t)(m)      * 32, 0, half), c0);
    c1 = wmma16(a, load_b_frag(bp + (size_t)(16 + m) * 32, 0, half), c1);
    c2 = wmma16(a, load_b_frag(bp + (size_t)(32 + m) * 32, 0, half), c2);
    c3 = wmma16(a, load_b_frag(bp + (size_t)(48 + m) * 32, 0, half), c3);
    __syncthreads();
    cur ^= 1;
  }

  float bv0 = bias[strip * 64 +      m];
  float bv1 = bias[strip * 64 + 16 + m];
  float bv2 = bias[strip * 64 + 32 + m];
  float bv3 = bias[strip * 64 + 48 + m];
#pragma unroll
  for (int r = 0; r < 8; ++r) {
    int row = w * 16 + r + 8 * half;
    tile[row][     m] = c0[r] + bv0;
    tile[row][16 + m] = c1[r] + bv1;
    tile[row][32 + m] = c2[r] + bv2;
    tile[row][48 + m] = c3[r] + bv3;
  }
  __syncthreads();

  int tid = threadIdx.x;
  if (s < 2 && tid < 64) {
    float su = 0.f, sq = 0.f;
    for (int cc = 0; cc < HD; ++cc) { float v = tile[tid][cc]; su += v; sq += v * v; }
    float mean = su * (1.f / HD);
    float var = sq * (1.f / HD) - mean * mean;
    s_mean[tid] = mean;
    s_rstd[tid] = rsqrtf(var + LN_EPS);
  }
  __syncthreads();

  if (s < 2) {
    const float* nw = (s == 0) ? qnw : knw;
    const float* nb = (s == 0) ? qnb : knb;
    float post = (s == 0) ? 0.125f : 1.f;   // fold q * HD^-0.5
    _Float16* dst = (s == 0) ? qbuf : kbuf;
#pragma unroll
    for (int i = 0; i < 32; ++i) {
      int e = tid * 32 + i;
      int row = e >> 6, cc = e & 63;
      float v = (tile[row][cc] - s_mean[row]) * s_rstd[row] * nw[cc] + nb[cc];
      int t = tt * 64 + row;
      int bb = t / SEQ, n = t - bb * SEQ;
      dst[(((size_t)bb * NH + h) * SEQ + n) * HD + cc] = (_Float16)(v * post);
    }
  } else {
#pragma unroll
    for (int i = 0; i < 32; ++i) {
      int e = tid * 32 + i;
      int row = e >> 6, cc = e & 63;
      int t = tt * 64 + row;
      int bb = t / SEQ, n = t - bb * SEQ;
      vtbuf[(((size_t)bb * NH + h) * HD + cc) * SEQ + n] = (_Float16)tile[row][cc];
    }
  }
}

// ---------------------------------------------------------------------------
// Flash attention. Block = 4 waves = 64 queries of one (b,h); per 32-key step
// the shared K(32x64)/VT(64x32) panels are TDM-staged (double buffered), each
// wave does 4 WMMA (S) -> online softmax -> P via LDS -> 4 WMMA (PV).
// ---------------------------------------------------------------------------
__global__ __launch_bounds__(128) void k_attn(const _Float16* __restrict__ qbuf,
                                              const _Float16* __restrict__ kbuf,
                                              const _Float16* __restrict__ vtbuf,
                                              const float* __restrict__ gates,
                                              _Float16* __restrict__ og) {
  __shared__ __align__(16) _Float16 sK[2][32 * 64];  // [key][d]
  __shared__ __align__(16) _Float16 sV[2][64 * 32];  // [d][key]
  __shared__ __align__(16) _Float16 sP[4][16 * 32];
  int w = threadIdx.x >> 5, lane = threadIdx.x & 31;
  int m = lane & 15, half = lane >> 4;
  int bh = blockIdx.x >> 5;           // b*12+h
  int qblk = blockIdx.x & 31;         // 64-query block
  int bb = bh / NH, h = bh - bb * NH;
  int qbase = qblk * 64 + w * 16;

  const _Float16* Q  = qbuf  + (size_t)bh * SEQ * HD;
  const _Float16* K  = kbuf  + (size_t)bh * SEQ * HD;
  const _Float16* VT = vtbuf + (size_t)bh * HD * SEQ;

  const _Float16* qrow = Q + (size_t)(qbase + m) * HD;
  v16h qf0 = load_a_frag(qrow, 0, half);
  v16h qf1 = load_a_frag(qrow, 32, half);

  v8f o0 = {0.f,0.f,0.f,0.f,0.f,0.f,0.f,0.f};
  v8f o1 = o0, o2 = o0, o3 = o0;
  float mrow[8], lrow[8];
#pragma unroll
  for (int r = 0; r < 8; ++r) { mrow[r] = -__builtin_inff(); lrow[r] = 0.f; }

  unsigned k0 = lds_off(&sK[0][0]), k1 = lds_off(&sK[1][0]);
  unsigned v0 = lds_off(&sV[0][0]), v1 = lds_off(&sV[1][0]);
  _Float16* myP = &sP[w][0];

  if (w == 0) {
    tdm_load_2d_f16(k0, K, 64, 32, HD);         // 32 key rows x 64 d
    tdm_load_2d_f16(v0, VT, 32, 64, SEQ);       // 64 d rows x 32 keys
  }
  int cur = 0;
  for (int j = 0; j < SEQ; j += 32) {
    if (w == 0) {
      if (j + 32 < SEQ) {
        tdm_load_2d_f16(cur ? k0 : k1, K + (size_t)(j + 32) * HD, 64, 32, HD);
        tdm_load_2d_f16(cur ? v0 : v1, VT + (j + 32), 32, 64, SEQ);
        __builtin_amdgcn_s_wait_tensorcnt(2);
      } else {
        __builtin_amdgcn_s_wait_tensorcnt(0);
      }
    }
    __syncthreads();
    const _Float16* Kt = &sK[cur][0];
    const _Float16* Vt = &sV[cur][0];

    // --- scores ---
    v8f s0 = {0.f,0.f,0.f,0.f,0.f,0.f,0.f,0.f};
    v8f s1 = s0;
    const _Float16* kr0 = Kt + (size_t)(m)      * HD;
    const _Float16* kr1 = Kt + (size_t)(16 + m) * HD;
    s0 = wmma16(qf0, load_b_frag(kr0, 0, half), s0);
    s0 = wmma16(qf1, load_b_frag(kr0, 32, half), s0);
    s1 = wmma16(qf0, load_b_frag(kr1, 0, half), s1);
    s1 = wmma16(qf1, load_b_frag(kr1, 32, half), s1);

    // --- online softmax (rows live in 16-lane half-groups) ---
#pragma unroll
    for (int r = 0; r < 8; ++r) {
      float a0 = s0[r], a1 = s1[r];
      float rmax = red_max16(fmaxf(a0, a1));
      float mnew = fmaxf(mrow[r], rmax);
      float alpha = __expf(mrow[r] - mnew);
      float p0 = __expf(a0 - mnew);
      float p1 = __expf(a1 - mnew);
      lrow[r] = lrow[r] * alpha + red_sum16(p0 + p1);
      mrow[r] = mnew;
      o0[r] *= alpha; o1[r] *= alpha; o2[r] *= alpha; o3[r] *= alpha;
      myP[(r + 8 * half) * 32 + m]      = (_Float16)p0;
      myP[(r + 8 * half) * 32 + 16 + m] = (_Float16)p1;
    }

    // --- P(16x32) @ V(32x64) ---
    v16h pf = load_a_frag(myP + m * 32, 0, half);
    o0 = wmma16(pf, load_b_frag(Vt + (size_t)(m)      * 32, 0, half), o0);
    o1 = wmma16(pf, load_b_frag(Vt + (size_t)(16 + m) * 32, 0, half), o1);
    o2 = wmma16(pf, load_b_frag(Vt + (size_t)(32 + m) * 32, 0, half), o2);
    o3 = wmma16(pf, load_b_frag(Vt + (size_t)(48 + m) * 32, 0, half), o3);
    __syncthreads();
    cur ^= 1;
  }

  // --- epilogue: normalize, apply MoH gate, store gated f16 activation ---
#pragma unroll
  for (int r = 0; r < 8; ++r) {
    int row = r + 8 * half;
    int t = bb * SEQ + qblk * 64 + w * 16 + row;
    float g = gates[(size_t)t * NH + h];
    float inv = g / lrow[r];
    _Float16* dst = og + (size_t)t * DIM + h * HD;
    dst[m]      = (_Float16)(o0[r] * inv);
    dst[16 + m] = (_Float16)(o1[r] * inv);
    dst[32 + m] = (_Float16)(o2[r] * inv);
    dst[48 + m] = (_Float16)(o3[r] * inv);
  }
}

// ---------------------------------------------------------------------------
// Output projection: out[8192][768] = og @ proj_w + proj_b (fp32 out).
// Same TDM-staged panel scheme; wave = 16 tokens x 64 cols.
// ---------------------------------------------------------------------------
__global__ __launch_bounds__(128) void k_proj(const _Float16* __restrict__ og,
                                              const _Float16* __restrict__ wT,
                                              const float* __restrict__ bias,
                                              float* __restrict__ out) {
  __shared__ __align__(16) _Float16 sB[2][64 * 32];
  int tt = blockIdx.x;          // 64-token tile
  int cs = blockIdx.y;          // 64-col strip
  int w = threadIdx.x >> 5, lane = threadIdx.x & 31;
  int m = lane & 15, half = lane >> 4;

  const _Float16* arow = og + (size_t)(tt * 64 + w * 16 + m) * DIM;
  const _Float16* wbase = wT + (size_t)(cs * 64) * DIM;
  unsigned l0 = lds_off(&sB[0][0]), l1 = lds_off(&sB[1][0]);

  v8f c0 = {0.f,0.f,0.f,0.f,0.f,0.f,0.f,0.f};
  v8f c1 = c0, c2 = c0, c3 = c0;

  if (w == 0) tdm_load_2d_f16(l0, wbase, 32, 64, DIM);
  int cur = 0;
  for (int kb = 0; kb < DIM; kb += 32) {
    if (w == 0) {
      if (kb + 32 < DIM) {
        tdm_load_2d_f16(cur ? l0 : l1, wbase + kb + 32, 32, 64, DIM);
        __builtin_amdgcn_s_wait_tensorcnt(1);
      } else {
        __builtin_amdgcn_s_wait_tensorcnt(0);
      }
    }
    __syncthreads();
    const _Float16* bp = &sB[cur][0];
    v16h a = load_a_frag(arow, kb, half);
    c0 = wmma16(a, load_b_frag(bp + (size_t)(m)      * 32, 0, half), c0);
    c1 = wmma16(a, load_b_frag(bp + (size_t)(16 + m) * 32, 0, half), c1);
    c2 = wmma16(a, load_b_frag(bp + (size_t)(32 + m) * 32, 0, half), c2);
    c3 = wmma16(a, load_b_frag(bp + (size_t)(48 + m) * 32, 0, half), c3);
    __syncthreads();
    cur ^= 1;
  }

#pragma unroll
  for (int r = 0; r < 8; ++r) {
    int t = tt * 64 + w * 16 + r + 8 * half;
    out[(size_t)t * DIM + cs * 64 +      m] = c0[r] + bias[cs * 64 +      m];
    out[(size_t)t * DIM + cs * 64 + 16 + m] = c1[r] + bias[cs * 64 + 16 + m];
    out[(size_t)t * DIM + cs * 64 + 32 + m] = c2[r] + bias[cs * 64 + 32 + m];
    out[(size_t)t * DIM + cs * 64 + 48 + m] = c3[r] + bias[cs * 64 + 48 + m];
  }
}

// ---------------------------------------------------------------------------
extern "C" void kernel_launch(void* const* d_in, const int* in_sizes, int n_in,
                              void* d_out, int out_size, void* d_ws, size_t ws_size,
                              hipStream_t stream) {
  const float* x        = (const float*)d_in[0];
  const float* qkv_w    = (const float*)d_in[1];
  const float* qkv_b    = (const float*)d_in[2];
  const float* q_norm_w = (const float*)d_in[3];
  const float* q_norm_b = (const float*)d_in[4];
  const float* k_norm_w = (const float*)d_in[5];
  const float* k_norm_b = (const float*)d_in[6];
  const float* wg_w     = (const float*)d_in[7];
  const float* wg0_w    = (const float*)d_in[8];
  const float* wg1_w    = (const float*)d_in[9];
  const float* proj_w   = (const float*)d_in[10];
  const float* proj_b   = (const float*)d_in[11];
  float* out = (float*)d_out;

  char* ws = (char*)d_ws;
  size_t off = 0;
  float*     gates = (float*)(ws + off);     off += (size_t)NTOK * NH * sizeof(float);
  _Float16*  xh    = (_Float16*)(ws + off);  off += (size_t)NTOK * DIM * sizeof(_Float16);
  _Float16*  qkvT  = (_Float16*)(ws + off);  off += (size_t)3 * DIM * DIM * sizeof(_Float16);
  _Float16*  projT = (_Float16*)(ws + off);  off += (size_t)DIM * DIM * sizeof(_Float16);
  _Float16*  qb    = (_Float16*)(ws + off);  off += (size_t)NTOK * DIM * sizeof(_Float16);
  _Float16*  kb    = (_Float16*)(ws + off);  off += (size_t)NTOK * DIM * sizeof(_Float16);
  _Float16*  vt    = (_Float16*)(ws + off);  off += (size_t)NTOK * DIM * sizeof(_Float16);
  _Float16*  og    = xh;   // x_f16 dead after k_qkv; reuse for gated attention out

  int nx = NTOK * DIM;
  k_cvt_x<<<(nx + 255) / 256, 256, 0, stream>>>(x, xh, nx);
  k_transpose_h<<<(DIM * 3 * DIM + 255) / 256, 256, 0, stream>>>(qkv_w, qkvT, DIM, 3 * DIM);
  k_transpose_h<<<(DIM * DIM + 255) / 256, 256, 0, stream>>>(proj_w, projT, DIM, DIM);
  k_gating<<<NTOK / 8, 256, 0, stream>>>(x, wg_w, wg0_w, wg1_w, gates);

  k_qkv<<<dim3(NTOK / 64, 36), 128, 0, stream>>>(xh, qkvT, qkv_b,
                                                 q_norm_w, q_norm_b, k_norm_w, k_norm_b,
                                                 qb, kb, vt);

  k_attn<<<BATCH * NH * (SEQ / 64), 128, 0, stream>>>(qb, kb, vt, gates, og);

  k_proj<<<dim3(NTOK / 64, DIM / 64), 128, 0, stream>>>(og, projT, proj_b, out);
}
